// MambaBlock_9440338116665
// MI455X (gfx1250) — matrix operand order repaired and checked by reading, not compiled, not run on hardware
//
#include <hip/hip_runtime.h>

// ---------------------------------------------------------------- constants
#define D_MODEL 1024
#define D_INNER 2048
#define D_STATE 16
#define DT_RANK 256
#define BATCH   2
#define SEQLEN  2048
#define BL      (BATCH * SEQLEN)            // 4096 rows
#define XPROJ_N (DT_RANK + 2 * D_STATE)     // 288

// Async global->LDS path (gfx1250): use if the toolchain declares the builtins.
#if defined(__AMDGCN__) && __has_builtin(__builtin_amdgcn_global_load_async_to_lds_b128) && __has_builtin(__builtin_amdgcn_s_wait_asynccnt)
#define USE_ASYNC_LDS 1
#else
#define USE_ASYNC_LDS 0
#endif

// ---------------------------------------------------------------- types
typedef __attribute__((ext_vector_type(8)))  float  v8f;
typedef __attribute__((ext_vector_type(16))) __bf16 v16bf;
typedef __attribute__((ext_vector_type(4)))  int    v4i;
#if USE_ASYNC_LDS
typedef __attribute__((address_space(1))) v4i AS1_v4i;   // global int4
typedef __attribute__((address_space(3))) v4i AS3_v4i;   // LDS int4
#endif

struct U128 { unsigned int x, y, z, w; };
struct U256 { U128 lo, hi; };

static __device__ __forceinline__ unsigned short f2bf(float f) {
  union { float f; unsigned int u; } c; c.f = f;
  unsigned int u = c.u;
  unsigned int r = u + 0x7FFFu + ((u >> 16) & 1u);   // round-to-nearest-even
  return (unsigned short)(r >> 16);
}
static __device__ __forceinline__ float bf2f(unsigned short v) {
  union { unsigned int u; float f; } c; c.u = ((unsigned int)v) << 16;
  return c.f;
}
static __device__ __forceinline__ v16bf ld_frag(const unsigned short* plo,
                                                const unsigned short* phi) {
  U256 t;
  t.lo = *reinterpret_cast<const U128*>(plo);
  t.hi = *reinterpret_cast<const U128*>(phi);
  return __builtin_bit_cast(v16bf, t);
}

// ---------------------------------------------------------------- elementwise
__global__ void k_f32_to_bf16(const float* __restrict__ in,
                              unsigned short* __restrict__ out, int n) {
  int i = blockIdx.x * 256 + threadIdx.x;
  if (i < n) out[i] = f2bf(in[i]);
}

// LayerNorm over last dim (1024), one block per row, bf16 output.
__global__ __launch_bounds__(256) void k_layernorm(const float* __restrict__ x,
    const float* __restrict__ w, const float* __restrict__ b,
    unsigned short* __restrict__ out) {
  __shared__ float red[256];
  const int row = blockIdx.x;
  const float* xr = x + (size_t)row * D_MODEL;
  const int t = threadIdx.x;
  float s = 0.f;
  for (int i = t; i < D_MODEL; i += 256) s += xr[i];
  red[t] = s; __syncthreads();
  for (int o = 128; o > 0; o >>= 1) { if (t < o) red[t] += red[t + o]; __syncthreads(); }
  float mu = red[0] * (1.f / D_MODEL);
  __syncthreads();
  float v = 0.f;
  for (int i = t; i < D_MODEL; i += 256) { float d = xr[i] - mu; v += d * d; }
  red[t] = v; __syncthreads();
  for (int o = 128; o > 0; o >>= 1) { if (t < o) red[t] += red[t + o]; __syncthreads(); }
  float rstd = rsqrtf(red[0] * (1.f / D_MODEL) + 1e-5f);
  for (int i = t; i < D_MODEL; i += 256) {
    float y = (xr[i] - mu) * rstd * w[i] + b[i];
    out[(size_t)row * D_MODEL + i] = f2bf(y);
  }
}

// ---------------------------------------------------------------- WMMA GEMM
// C[M,N] (f32) = A[M,K] (bf16, row-major) * B[K,N] (bf16, row-major)
// Block tile 128x128, K-step 32, double-buffered LDS. 8 waves:
// wm=wave>>1 owns a 32-row band, wn=wave&1 owns a 64-col band ->
// 2x4 accumulators (16x16) per wave, 8 v_wmma per K-step.
// Tile i+1 is staged (async-DMA A + transpose-store B) before computing
// tile i; one s_wait_asynccnt + one barrier per K-step.
__global__ __launch_bounds__(256) void k_gemm_bf16(
    const unsigned short* __restrict__ A, const unsigned short* __restrict__ B,
    float* __restrict__ C, int M, int N, int K) {
  __shared__ unsigned short As[2][128][32];  // [buf][m][k]   2 x 8 KB
  __shared__ unsigned short Bs[2][128][32];  // [buf][n][k]   2 x 8 KB (transposed)
  const int tid   = threadIdx.x;
  const int wave  = tid >> 5;
  const int lane  = tid & 31;
  const int wm    = wave >> 1;        // 0..3  (32-row band)
  const int wn    = wave & 1;         // 0..1  (64-col band)
  const int lhalf = lane >> 4;        // 0/1 (lane group)
  const int l15   = lane & 15;
  const int m0 = blockIdx.y * 128;
  const int n0 = blockIdx.x * 128;
  const int ar = tid >> 2, ac = (tid & 3) * 8;   // A loader: rows ar, ar+64
  const int bk = tid >> 3, bc8 = (tid & 7) * 8;  // B loader: row bk, cols bc8, bc8+64
  const bool fullN = (n0 + 128 <= N);

  v8f acc[2][4];
  #pragma unroll
  for (int i = 0; i < 2; ++i)
    #pragma unroll
    for (int t = 0; t < 4; ++t)
      acc[i][t] = (v8f){0.f, 0.f, 0.f, 0.f, 0.f, 0.f, 0.f, 0.f};

  auto stageA = [&](int k0, int buf) {
#if USE_ASYNC_LDS
    __builtin_amdgcn_global_load_async_to_lds_b128(
        (AS1_v4i*)(void*)(A + (size_t)(m0 + ar) * K + k0 + ac),
        (AS3_v4i*)(void*)&As[buf][ar][ac], 0, 0);
    __builtin_amdgcn_global_load_async_to_lds_b128(
        (AS1_v4i*)(void*)(A + (size_t)(m0 + ar + 64) * K + k0 + ac),
        (AS3_v4i*)(void*)&As[buf][ar + 64][ac], 0, 0);
#else
    *reinterpret_cast<U128*>(&As[buf][ar][ac]) =
        *reinterpret_cast<const U128*>(A + (size_t)(m0 + ar) * K + k0 + ac);
    *reinterpret_cast<U128*>(&As[buf][ar + 64][ac]) =
        *reinterpret_cast<const U128*>(A + (size_t)(m0 + ar + 64) * K + k0 + ac);
#endif
  };
  auto stageB = [&](int k0, int buf) {
    if (fullN) {
      #pragma unroll
      for (int h = 0; h < 2; ++h) {
        U128 v = *reinterpret_cast<const U128*>(
            B + (size_t)(k0 + bk) * N + n0 + h * 64 + bc8);
        const unsigned short* pv = reinterpret_cast<const unsigned short*>(&v);
        #pragma unroll
        for (int j = 0; j < 8; ++j) Bs[buf][h * 64 + bc8 + j][bk] = pv[j];
      }
    } else {
      // branch-free edge: clamp address, select 0 past N (keeps EXEC uniform)
      #pragma unroll
      for (int h = 0; h < 2; ++h)
        #pragma unroll
        for (int j = 0; j < 8; ++j) {
          int n = n0 + h * 64 + bc8 + j;
          int nc = (n < N) ? n : (N - 1);
          unsigned short v = B[(size_t)(k0 + bk) * N + nc];
          Bs[buf][h * 64 + bc8 + j][bk] = (n < N) ? v : (unsigned short)0;
        }
    }
  };

  // prologue: stage tile 0 into buffer 0
  stageA(0, 0);
  stageB(0, 0);
#if USE_ASYNC_LDS
  __builtin_amdgcn_s_wait_asynccnt(0);
#endif
  __syncthreads();

  const int nk = K >> 5;
  for (int it = 0; it < nk; ++it) {
    const int cur = it & 1;
    if (it + 1 < nk) {           // stage next tile into the other buffer
      stageA((it + 1) << 5, cur ^ 1);
      stageB((it + 1) << 5, cur ^ 1);
    }
    // A fragments: lane row m; halves 0..7 = K kb..kb+7, halves 8..15 =
    // K 16+kb..16+kb+7, kb = lhalf*8  (ISA 16-bit A 16x32 layout)
    const unsigned short* arow0 = &As[cur][wm * 32 + l15][lhalf * 8];
    const unsigned short* arow1 = &As[cur][wm * 32 + 16 + l15][lhalf * 8];
    v16bf a0 = ld_frag(arow0, arow0 + 16);
    v16bf a1 = ld_frag(arow1, arow1 + 16);
    // B fragments: lane col n = base + l15; same K order as A (kb..kb+7, 16+kb..16+kb+7)
    #pragma unroll
    for (int t = 0; t < 4; ++t) {
      const unsigned short* brow = &Bs[cur][wn * 64 + t * 16 + l15][lhalf * 8];
      v16bf bf = ld_frag(brow, brow + 16);
      acc[0][t] = __builtin_amdgcn_wmma_f32_16x16x32_bf16(
          false, a0, false, bf, (short)0, acc[0][t], false, false);
      acc[1][t] = __builtin_amdgcn_wmma_f32_16x16x32_bf16(
          false, a1, false, bf, (short)0, acc[1][t], false, false);
    }
#if USE_ASYNC_LDS
    __builtin_amdgcn_s_wait_asynccnt(0);   // next-buffer A DMA complete
#endif
    __syncthreads();                       // next-buffer B stores visible
  }

  // C/D layout: VGPR j -> M = j + 8*lhalf, N = l15 (ISA 32-bit C/D 16x16)
  const int rb = m0 + wm * 32 + lhalf * 8;
  const int cb = n0 + wn * 64 + l15;
  #pragma unroll
  for (int i = 0; i < 2; ++i)
    #pragma unroll
    for (int t = 0; t < 4; ++t) {
      const int c = cb + t * 16;
      if (c < N) {
        #pragma unroll
        for (int j = 0; j < 8; ++j)
          C[(size_t)(rb + i * 16 + j) * N + c] = acc[i][t][j];
      }
    }
}

// ---------------------------------------------------------------- epilogues
// u = silu(xz[:, :2048]) -> bf16
__global__ void k_silu_u(const float* __restrict__ xz,
                         unsigned short* __restrict__ u_bf) {
  int i = blockIdx.x * 256 + threadIdx.x;          // over BL*D_INNER
  int row = i >> 11, col = i & (D_INNER - 1);
  float v = xz[((size_t)row << 12) + col];
  float u = v / (1.f + __expf(-v));
  u_bf[i] = f2bf(u);
}

// delta_raw = x_dbl[:, :256] -> bf16 (compact)
__global__ void k_dtraw(const float* __restrict__ xdbl,
                        unsigned short* __restrict__ d_bf) {
  int i = blockIdx.x * 256 + threadIdx.x;          // over BL*DT_RANK
  int row = i >> 8, col = i & (DT_RANK - 1);
  d_bf[i] = f2bf(xdbl[(size_t)row * XPROJ_N + col]);
}

// delta = softplus(dt_lin + bias), in place
__global__ void k_softplus(float* __restrict__ dt, const float* __restrict__ bias) {
  int i = blockIdx.x * 256 + threadIdx.x;          // over BL*D_INNER
  float x = dt[i] + bias[i & (D_INNER - 1)];
  dt[i] = (x > 20.f) ? x : log1pf(__expf(x));
}

// ---------------------------------------------------------------- selective scan
// one thread per (b, d) channel; h[16] in registers; B/C staged via LDS.
__global__ __launch_bounds__(256) void k_scan(const float* __restrict__ delta,
    const unsigned short* __restrict__ u_bf, const float* __restrict__ xdbl,
    const float* __restrict__ xz, const float* __restrict__ A_log,
    const float* __restrict__ Dvec, unsigned short* __restrict__ y_bf) {
  __shared__ float bcs[64 * 32];                   // 64 steps x (B[16], C[16])
  const int d = blockIdx.x * 256 + threadIdx.x;    // channel
  const int b = blockIdx.y;                        // batch
  float A[D_STATE], h[D_STATE];
  #pragma unroll
  for (int n = 0; n < D_STATE; ++n) {
    A[n] = -__expf(A_log[(size_t)d * D_STATE + n]);
    h[n] = 0.f;
  }
  const float Dd = Dvec[d];
  for (int tb = 0; tb < SEQLEN; tb += 64) {
    __syncthreads();
    for (int idx = threadIdx.x; idx < 64 * 32; idx += 256) {
      int tt = idx >> 5, c = idx & 31;
      bcs[idx] = xdbl[(size_t)(b * SEQLEN + tb + tt) * XPROJ_N + DT_RANK + c];
    }
    __syncthreads();
    for (int tt = 0; tt < 64; ++tt) {
      const size_t ti = (size_t)(b * SEQLEN + tb + tt);
      const float dt = delta[ti * D_INNER + d];
      const float u  = bf2f(u_bf[ti * D_INNER + d]);
      const float* Bt = &bcs[tt * 32];
      const float* Ct = Bt + D_STATE;
      const float du = dt * u;
      float y = 0.f;
      #pragma unroll
      for (int n = 0; n < D_STATE; ++n) {
        h[n] = __expf(dt * A[n]) * h[n] + du * Bt[n];
        y += h[n] * Ct[n];
      }
      const float z  = xz[ti * (2 * D_INNER) + D_INNER + d];
      const float yy = (y + u * Dd) * (z / (1.f + __expf(-z)));
      y_bf[ti * D_INNER + d] = f2bf(yy);
    }
  }
}

// out += residual x
__global__ void k_residual(float* __restrict__ out, const float* __restrict__ x) {
  int i = blockIdx.x * 256 + threadIdx.x;
  out[i] += x[i];
}

// ---------------------------------------------------------------- launcher
extern "C" void kernel_launch(void* const* d_in, const int* in_sizes, int n_in,
                              void* d_out, int out_size, void* d_ws, size_t ws_size,
                              hipStream_t stream) {
  (void)in_sizes; (void)n_in; (void)out_size; (void)ws_size;
  const float* x        = (const float*)d_in[0];
  const float* norm_w   = (const float*)d_in[1];
  const float* norm_b   = (const float*)d_in[2];
  const float* in_proj  = (const float*)d_in[3];
  const float* x_proj   = (const float*)d_in[4];
  const float* dt_proj  = (const float*)d_in[5];
  const float* dt_bias  = (const float*)d_in[6];
  const float* A_log    = (const float*)d_in[7];
  const float* Dvec     = (const float*)d_in[8];
  const float* out_proj = (const float*)d_in[9];
  float* out = (float*)d_out;

  char* p = (char*)d_ws;
  auto take = [&](size_t bytes) {
    char* r = p;
    p += (bytes + 255) & ~(size_t)255;
    return r;
  };
  unsigned short* inW  = (unsigned short*)take((size_t)D_MODEL * 2 * D_INNER * 2);
  unsigned short* xW   = (unsigned short*)take((size_t)D_INNER * XPROJ_N * 2);
  unsigned short* dtW  = (unsigned short*)take((size_t)DT_RANK * D_INNER * 2);
  unsigned short* oW   = (unsigned short*)take((size_t)D_INNER * D_MODEL * 2);
  unsigned short* xnB  = (unsigned short*)take((size_t)BL * D_MODEL * 2);
  float*          xz   = (float*)take((size_t)BL * 2 * D_INNER * 4);
  unsigned short* uB   = (unsigned short*)take((size_t)BL * D_INNER * 2);
  float*          xdbl = (float*)take((size_t)BL * XPROJ_N * 4);
  unsigned short* drB  = (unsigned short*)take((size_t)BL * DT_RANK * 2);
  float*          dlt  = (float*)take((size_t)BL * D_INNER * 4);
  unsigned short* yB   = (unsigned short*)take((size_t)BL * D_INNER * 2);

  auto cdiv = [](int a, int b) { return (a + b - 1) / b; };

  { int n = D_MODEL * 2 * D_INNER; k_f32_to_bf16<<<cdiv(n, 256), 256, 0, stream>>>(in_proj, inW, n); }
  { int n = D_INNER * XPROJ_N;     k_f32_to_bf16<<<cdiv(n, 256), 256, 0, stream>>>(x_proj, xW, n); }
  { int n = DT_RANK * D_INNER;     k_f32_to_bf16<<<cdiv(n, 256), 256, 0, stream>>>(dt_proj, dtW, n); }
  { int n = D_INNER * D_MODEL;     k_f32_to_bf16<<<cdiv(n, 256), 256, 0, stream>>>(out_proj, oW, n); }

  k_layernorm<<<BL, 256, 0, stream>>>(x, norm_w, norm_b, xnB);

  // xz = xn @ in_proj   [4096 x 4096], K=1024
  k_gemm_bf16<<<dim3(cdiv(2 * D_INNER, 128), BL / 128), 256, 0, stream>>>(
      xnB, inW, xz, BL, 2 * D_INNER, D_MODEL);

  k_silu_u<<<(BL * D_INNER) / 256, 256, 0, stream>>>(xz, uB);

  // x_dbl = u @ x_proj  [4096 x 288], K=2048
  k_gemm_bf16<<<dim3(cdiv(XPROJ_N, 128), BL / 128), 256, 0, stream>>>(
      uB, xW, xdbl, BL, XPROJ_N, D_INNER);

  k_dtraw<<<(BL * DT_RANK) / 256, 256, 0, stream>>>(xdbl, drB);

  // dt_lin = delta_raw @ dt_proj  [4096 x 2048], K=256
  k_gemm_bf16<<<dim3(cdiv(D_INNER, 128), BL / 128), 256, 0, stream>>>(
      drB, dtW, dlt, BL, D_INNER, DT_RANK);

  k_softplus<<<(BL * D_INNER) / 256, 256, 0, stream>>>(dlt, dt_bias);

  k_scan<<<dim3(D_INNER / 256, BATCH), 256, 0, stream>>>(
      dlt, uB, xdbl, xz, A_log, Dvec, yB);

  // out = y @ out_proj  [4096 x 1024], K=2048, then += residual
  k_gemm_bf16<<<dim3(cdiv(D_MODEL, 128), BL / 128), 256, 0, stream>>>(
      yB, oW, out, BL, D_MODEL, D_INNER);

  k_residual<<<(BL * D_MODEL) / 256, 256, 0, stream>>>(out, x);
}
